// T2TGRU_67319317397521
// MI455X (gfx1250) — matrix operand order repaired and checked by reading, not compiled, not run on hardware
//
#include <hip/hip_runtime.h>
#include <hip/hip_bf16.h>

// ---------------------------------------------------------------------------
// Types for CDNA5 WMMA (wave32): v_wmma_f32_16x16x32_bf16
// ---------------------------------------------------------------------------
typedef __attribute__((ext_vector_type(16))) __bf16 v16bf;
typedef __attribute__((ext_vector_type(8)))  float  v8f;

#define FEATURE_DIM 320
#define ASPECT_OUT  64
#define HIDDEN      512
#define CDIM        576      // 64 + 512 (concat dim, K of recurrent GEMMs)
#define BATCH       2048
#define SEQ         256
#define BT          32       // batch rows per workgroup in the GRU kernel
#define XH_STRIDE   584      // 576 + 8 pad (bf16 elements) -> conflict-free rows
#define RH_STRIDE   520      // 512 + 8 pad

// ---------------------------------------------------------------------------
// bf16 helpers: native casts -> v_cvt_pk_bf16_f32 (confirmed in round-2 asm)
// ---------------------------------------------------------------------------
static __device__ __forceinline__ unsigned short f2bf_u16(float f) {
  return __builtin_bit_cast(unsigned short, (__bf16)f);
}
static __device__ __forceinline__ float bf2f(unsigned short s) {
  return (float)__builtin_bit_cast(__bf16, s);
}
static __device__ __forceinline__ v8f splat8(float b) {
  v8f v = {b, b, b, b, b, b, b, b};
  return v;
}

union BfFrag {
  uint4 q[2];
  v16bf v;
  unsigned short s[16];
  __bf16 h[16];
};

// B fragment: (K=32 x N=16) bf16, pre-swizzled so each lane reads 32 contiguous
// bytes. Lanes 0-15 = column n, K 0..7,16..23; lanes 16-31 =
// column n, K 8..15,24..31 (matches ISA 16-bit B operand layout).
static __device__ __forceinline__ v16bf load_frag32(const unsigned short* p) {
  BfFrag u;
  u.q[0] = *(const uint4*)(p);
  u.q[1] = *(const uint4*)(p + 8);
  return u.v;
}
static __device__ __forceinline__ v16bf load_b(const unsigned short* w, int frag, int lane) {
  return load_frag32(w + ((size_t)frag * 32 + lane) * 16);
}

// A fragment: (M=16 x K=32) bf16 from row-major bf16 matrix in LDS.
// ISA layout: lane L holds row L%16; lanes 0-15 hold K [k0..k0+7],[k0+16..k0+23],
// lanes 16-31 hold K [k0+8..k0+15],[k0+24..k0+31].
static __device__ __forceinline__ v16bf load_a(const unsigned short* base, int stride,
                                               int rowbase, int k0, int lane) {
  int row  = rowbase + (lane & 15);
  int koff = (lane >> 4) * 8;
  const unsigned short* p = base + row * stride + k0 + koff;
  BfFrag u;
  u.q[0] = *(const uint4*)(p);
  u.q[1] = *(const uint4*)(p + 16);
  return u.v;
}

static __device__ __forceinline__ v8f wmma_bf16(v16bf a, v16bf b, v8f c) {
  return __builtin_amdgcn_wmma_f32_16x16x32_bf16(false, a, false, b, (short)0, c, false, false);
}

static __device__ __forceinline__ float sigmoidf_(float x) {
  return 1.f / (1.f + __expf(-x));
}
static __device__ __forceinline__ float tanhf_(float x) {
  x = fminf(fmaxf(x, -15.f), 15.f);
  float e = __expf(2.f * x);
  return (e - 1.f) / (e + 1.f);
}

// ---------------------------------------------------------------------------
// Kernel 1: scatter the 4 small aspect weights into a dense (320 x 64) f32
// matrix (row-major K x N). dense must be pre-zeroed.
// ---------------------------------------------------------------------------
__global__ void aspect_scatter(const float* __restrict__ woff, const float* __restrict__ wdef,
                               const float* __restrict__ werr, const float* __restrict__ winf,
                               float* __restrict__ dense) {
  int t = blockIdx.x * blockDim.x + threadIdx.x;
  const int off5[5] = {2, 5, 8, 9, 12};
  const int off3[3] = {10, 14, 15};
  const int off2e[2] = {13, 17};
  const int off2i[2] = {18, 19};
  if (t < 1280) {                    // off: 16 x 80
    int a = t / 80, j = t % 80;
    int k = 20 * (j / 5) + off5[j % 5];
    dense[k * ASPECT_OUT + a] = woff[a * 80 + j];
  } else if (t < 2048) {             // def: 16 x 48
    int q = t - 1280; int a = q / 48, j = q % 48;
    int k = 20 * (j / 3) + off3[j % 3];
    dense[k * ASPECT_OUT + 16 + a] = wdef[a * 48 + j];
  } else if (t < 2560) {             // err: 16 x 32
    int q = t - 2048; int a = q / 32, j = q % 32;
    int k = 20 * (j / 2) + off2e[j % 2];
    dense[k * ASPECT_OUT + 32 + a] = werr[a * 32 + j];
  } else if (t < 3072) {             // inf: 16 x 32
    int q = t - 2560; int a = q / 32, j = q % 32;
    int k = 20 * (j / 2) + off2i[j % 2];
    dense[k * ASPECT_OUT + 48 + a] = winf[a * 32 + j];
  }
}

// ---------------------------------------------------------------------------
// Kernel 2: swizzle a row-major (K x N) f32 weight matrix into bf16 WMMA
// B-fragments, fragment id = kt*(N/16)+nt, per-lane 16 contiguous bf16.
// ---------------------------------------------------------------------------
__global__ void swizzle_kernel(const float* __restrict__ src, unsigned short* __restrict__ dst,
                               int K, int N) {
  int t = blockIdx.x * blockDim.x + threadIdx.x;
  int nfrag = (K >> 5) * (N >> 4);
  if (t >= nfrag * 32) return;
  int frag = t >> 5, lane = t & 31;
  int NT = N >> 4;
  int kt = frag / NT, nt = frag % NT;
  int n = nt * 16 + (lane & 15);
  int kbase = kt * 32 + ((lane >> 4) << 3);
  unsigned short* d = dst + ((size_t)frag * 32 + lane) * 16;
#pragma unroll
  for (int v = 0; v < 16; ++v)
    d[v] = f2bf_u16(src[(size_t)(kbase + v + (v >= 8 ? 8 : 0)) * N + n]);
}

// ---------------------------------------------------------------------------
// Kernel 3: aspect projection. rows = B*S, each wave handles a 16-row M-tile,
// N=64 (4 tiles), K=320 (10 steps, fully unrolled, B loads software-pipelined).
// ---------------------------------------------------------------------------
__global__ __launch_bounds__(256, 1) void proj_kernel(
    const float* __restrict__ inp, const unsigned short* __restrict__ asw,
    const float* __restrict__ boff, const float* __restrict__ bdef,
    const float* __restrict__ berr, const float* __restrict__ binf,
    unsigned short* __restrict__ xout) {
  __shared__ __align__(16) unsigned short s_w[40 * 512];  // 40 frags * 512 bf16 = 40 KB
  const int tid = threadIdx.x, lane = tid & 31, wave = tid >> 5;

  for (int i = tid * 8; i < 40 * 512; i += 256 * 8)
    *(uint4*)(&s_w[i]) = *(const uint4*)(&asw[i]);

  const int lcol = lane & 15;
  float bias[4];
  bias[0] = boff[lcol]; bias[1] = bdef[lcol]; bias[2] = berr[lcol]; bias[3] = binf[lcol];
  __syncthreads();

  const size_t row0 = ((size_t)blockIdx.x * 8 + wave) * 16;
  const int koff = (lane >> 4) * 8;
  const float* rp = inp + (row0 + lcol) * FEATURE_DIM;

  v8f acc[4];
#pragma unroll
  for (int nt = 0; nt < 4; ++nt) acc[nt] = splat8(bias[nt]);

#pragma unroll
  for (int kt = 0; kt < 10; ++kt) {
    const float* p = rp + kt * 32 + koff;
    float4 f0 = *(const float4*)(p);
    float4 f1 = *(const float4*)(p + 4);
    float4 f2 = *(const float4*)(p + 16);
    float4 f3 = *(const float4*)(p + 20);
    BfFrag fa;
    fa.h[0] = (__bf16)f0.x; fa.h[1] = (__bf16)f0.y; fa.h[2] = (__bf16)f0.z; fa.h[3] = (__bf16)f0.w;
    fa.h[4] = (__bf16)f1.x; fa.h[5] = (__bf16)f1.y; fa.h[6] = (__bf16)f1.z; fa.h[7] = (__bf16)f1.w;
    fa.h[8]  = (__bf16)f2.x; fa.h[9]  = (__bf16)f2.y; fa.h[10] = (__bf16)f2.z; fa.h[11] = (__bf16)f2.w;
    fa.h[12] = (__bf16)f3.x; fa.h[13] = (__bf16)f3.y; fa.h[14] = (__bf16)f3.z; fa.h[15] = (__bf16)f3.w;
    // software-pipelined B loads: each wait covers a load issued >=2 slots back
    v16bf b0 = load_b(s_w, kt * 4 + 0, lane);
    v16bf b1 = load_b(s_w, kt * 4 + 1, lane);
    acc[0] = wmma_bf16(fa.v, b0, acc[0]);
    v16bf b2 = load_b(s_w, kt * 4 + 2, lane);
    acc[1] = wmma_bf16(fa.v, b1, acc[1]);
    v16bf b3 = load_b(s_w, kt * 4 + 3, lane);
    acc[2] = wmma_bf16(fa.v, b2, acc[2]);
    acc[3] = wmma_bf16(fa.v, b3, acc[3]);
  }

  const int rowoff = (lane >> 4) * 8;  // C layout: VGPR i = row i (lo lanes) / 8+i (hi lanes)
#pragma unroll
  for (int nt = 0; nt < 4; ++nt)
#pragma unroll
    for (int i = 0; i < 8; ++i)
      xout[(row0 + rowoff + i) * 64 + nt * 16 + lcol] = f2bf_u16(acc[nt][i]);
}

// ---------------------------------------------------------------------------
// Kernel 4: persistent GRU. One WG per 32 batch rows, 8 waves. h lives in LDS
// for all 256 steps. Wave w owns gate/candidate columns [w*64, w*64+64).
// r and u gates computed in ONE K-pass (shared A fragments, 16 accumulators).
// Weights stream from L2 as pre-swizzled B-fragments, loads pipelined ahead
// of consuming WMMAs. Next step's x tile prefetched (global_prefetch_b8).
// ---------------------------------------------------------------------------
__global__ __launch_bounds__(256, 1) void gru_kernel(
    const unsigned short* __restrict__ xall,   // [B][S][64] bf16
    const unsigned short* __restrict__ w1sw,   // swizzled (576 x 1024)
    const unsigned short* __restrict__ w2sw,   // swizzled (576 x 512)
    const float* __restrict__ b1, const float* __restrict__ b2,
    float* __restrict__ out) {
  __shared__ __align__(16) unsigned short s_xh[BT * XH_STRIDE];  // cols 0..63 = x, 64..575 = h
  __shared__ __align__(16) unsigned short s_rh[BT * RH_STRIDE];  // r*h (32 x 512)

  const int tid = threadIdx.x;
  const int lane = tid & 31;
  const int wave = tid >> 5;
  const int b0 = blockIdx.x * BT;
  const int lcol = lane & 15;
  const int rowoff = (lane >> 4) * 8;

  for (int i = tid; i < BT * XH_STRIDE; i += 256) s_xh[i] = 0;  // h0 = 0

  float biasG[8], biasC[4];
#pragma unroll
  for (int nt = 0; nt < 4; ++nt) {
    int col = wave * 64 + nt * 16 + lcol;
    biasG[nt]     = b1[col];          // r-gate bias
    biasG[4 + nt] = b1[512 + col];    // u-gate bias
    biasC[nt]     = b2[col];
  }
  const int fragR = wave * 4;        // r-gate N-tile base in W1 (cols 0..511)
  const int fragU = 32 + wave * 4;   // u-gate N-tile base in W1 (cols 512..1023)
  const int fragC = wave * 4;        // candidate N-tile base in W2
  const int xrow = tid >> 3;         // x-staging: 8 threads per row
  const int xc8  = (tid & 7) * 8;
  __syncthreads();

  for (int s = 0; s < SEQ; ++s) {
    // ---- stage x_t tile (32 rows x 64 bf16) into LDS ----
    {
      uint4 v = *(const uint4*)(xall + ((size_t)(b0 + xrow) * SEQ + s) * 64 + xc8);
      *(uint4*)(&s_xh[xrow * XH_STRIDE + xc8]) = v;
    }
    // prefetch next step's tile into cache (global_prefetch_b8)
    if (s + 1 < SEQ)
      __builtin_prefetch(xall + ((size_t)(b0 + xrow) * SEQ + (s + 1)) * 64 + xc8, 0, 1);
    __syncthreads();

    // ---- phase RU: r and u gate slices in one pass over K ----
    v8f accG[2][8];
#pragma unroll
    for (int mt = 0; mt < 2; ++mt)
#pragma unroll
      for (int g = 0; g < 8; ++g) accG[mt][g] = splat8(biasG[g]);
#pragma unroll
    for (int kt = 0; kt < 18; ++kt) {
      v16bf a0 = load_a(s_xh, XH_STRIDE, 0,  kt * 32, lane);
      v16bf a1 = load_a(s_xh, XH_STRIDE, 16, kt * 32, lane);
      v16bf br0 = load_b(w1sw, kt * 64 + fragR + 0, lane);
      v16bf br1 = load_b(w1sw, kt * 64 + fragR + 1, lane);
      accG[0][0] = wmma_bf16(a0, br0, accG[0][0]);
      accG[1][0] = wmma_bf16(a1, br0, accG[1][0]);
      v16bf br2 = load_b(w1sw, kt * 64 + fragR + 2, lane);
      accG[0][1] = wmma_bf16(a0, br1, accG[0][1]);
      accG[1][1] = wmma_bf16(a1, br1, accG[1][1]);
      v16bf br3 = load_b(w1sw, kt * 64 + fragR + 3, lane);
      accG[0][2] = wmma_bf16(a0, br2, accG[0][2]);
      accG[1][2] = wmma_bf16(a1, br2, accG[1][2]);
      v16bf bu0 = load_b(w1sw, kt * 64 + fragU + 0, lane);
      accG[0][3] = wmma_bf16(a0, br3, accG[0][3]);
      accG[1][3] = wmma_bf16(a1, br3, accG[1][3]);
      v16bf bu1 = load_b(w1sw, kt * 64 + fragU + 1, lane);
      accG[0][4] = wmma_bf16(a0, bu0, accG[0][4]);
      accG[1][4] = wmma_bf16(a1, bu0, accG[1][4]);
      v16bf bu2 = load_b(w1sw, kt * 64 + fragU + 2, lane);
      accG[0][5] = wmma_bf16(a0, bu1, accG[0][5]);
      accG[1][5] = wmma_bf16(a1, bu1, accG[1][5]);
      v16bf bu3 = load_b(w1sw, kt * 64 + fragU + 3, lane);
      accG[0][6] = wmma_bf16(a0, bu2, accG[0][6]);
      accG[1][6] = wmma_bf16(a1, bu2, accG[1][6]);
      accG[0][7] = wmma_bf16(a0, bu3, accG[0][7]);
      accG[1][7] = wmma_bf16(a1, bu3, accG[1][7]);
    }
    // r -> r*h into s_rh; u -> sigmoid in registers
#pragma unroll
    for (int mt = 0; mt < 2; ++mt)
#pragma unroll
      for (int nt = 0; nt < 4; ++nt)
#pragma unroll
        for (int i = 0; i < 8; ++i) {
          int row = mt * 16 + rowoff + i;
          int col = wave * 64 + nt * 16 + lcol;
          float rg = sigmoidf_(accG[mt][nt][i]);
          float hv = bf2f(s_xh[row * XH_STRIDE + 64 + col]);
          s_rh[row * RH_STRIDE + col] = f2bf_u16(rg * hv);
          accG[mt][4 + nt][i] = sigmoidf_(accG[mt][4 + nt][i]);
        }

    __syncthreads();  // r*h complete for all waves

    // ---- phase C: candidate slice over concat(x, r*h) ----
    v8f accC[2][4];
#pragma unroll
    for (int mt = 0; mt < 2; ++mt)
#pragma unroll
      for (int nt = 0; nt < 4; ++nt) accC[mt][nt] = splat8(biasC[nt]);
#pragma unroll
    for (int kt = 0; kt < 18; ++kt) {
      v16bf a0, a1;
      if (kt < 2) {  // K 0..63 -> x region
        a0 = load_a(s_xh, XH_STRIDE, 0,  kt * 32, lane);
        a1 = load_a(s_xh, XH_STRIDE, 16, kt * 32, lane);
      } else {       // K 64..575 -> r*h
        a0 = load_a(s_rh, RH_STRIDE, 0,  (kt - 2) * 32, lane);
        a1 = load_a(s_rh, RH_STRIDE, 16, (kt - 2) * 32, lane);
      }
      v16bf c0 = load_b(w2sw, kt * 32 + fragC + 0, lane);
      v16bf c1 = load_b(w2sw, kt * 32 + fragC + 1, lane);
      accC[0][0] = wmma_bf16(a0, c0, accC[0][0]);
      accC[1][0] = wmma_bf16(a1, c0, accC[1][0]);
      v16bf c2 = load_b(w2sw, kt * 32 + fragC + 2, lane);
      accC[0][1] = wmma_bf16(a0, c1, accC[0][1]);
      accC[1][1] = wmma_bf16(a1, c1, accC[1][1]);
      v16bf c3 = load_b(w2sw, kt * 32 + fragC + 3, lane);
      accC[0][2] = wmma_bf16(a0, c2, accC[0][2]);
      accC[1][2] = wmma_bf16(a1, c2, accC[1][2]);
      accC[0][3] = wmma_bf16(a0, c3, accC[0][3]);
      accC[1][3] = wmma_bf16(a1, c3, accC[1][3]);
    }

    // ---- h_new = u*h + (1-u)*c ----
#pragma unroll
    for (int mt = 0; mt < 2; ++mt)
#pragma unroll
      for (int nt = 0; nt < 4; ++nt)
#pragma unroll
        for (int i = 0; i < 8; ++i) {
          int row = mt * 16 + rowoff + i;
          int col = wave * 64 + nt * 16 + lcol;
          float c = tanhf_(accC[mt][nt][i]);
          float u = accG[mt][4 + nt][i];
          float ho = bf2f(s_xh[row * XH_STRIDE + 64 + col]);
          s_xh[row * XH_STRIDE + 64 + col] = f2bf_u16(u * ho + (1.f - u) * c);
        }
    __syncthreads();
  }

  // write h_last (f32) coalesced
  for (int i = tid; i < BT * HIDDEN; i += 256) {
    int r = i >> 9, c = i & 511;
    out[(size_t)(b0 + r) * HIDDEN + c] = bf2f(s_xh[r * XH_STRIDE + 64 + c]);
  }
}

// ---------------------------------------------------------------------------
// Launch
// ---------------------------------------------------------------------------
extern "C" void kernel_launch(void* const* d_in, const int* in_sizes, int n_in,
                              void* d_out, int out_size, void* d_ws, size_t ws_size,
                              hipStream_t stream) {
  (void)in_sizes; (void)n_in; (void)out_size; (void)ws_size;
  const float* inputs = (const float*)d_in[0];
  const float* w_off  = (const float*)d_in[1];
  const float* b_off  = (const float*)d_in[2];
  const float* w_def  = (const float*)d_in[3];
  const float* b_def  = (const float*)d_in[4];
  const float* w_err  = (const float*)d_in[5];
  const float* b_err  = (const float*)d_in[6];
  const float* w_inf  = (const float*)d_in[7];
  const float* b_inf  = (const float*)d_in[8];
  const float* W1     = (const float*)d_in[9];
  const float* b1     = (const float*)d_in[10];
  const float* W2     = (const float*)d_in[11];
  const float* b2     = (const float*)d_in[12];

  char* ws = (char*)d_ws;
  size_t o = 0;
  unsigned short* xbuf = (unsigned short*)(ws + o); o += (size_t)BATCH * SEQ * 64 * 2;       // 67 MB
  unsigned short* w1sw = (unsigned short*)(ws + o); o += (size_t)CDIM * 1024 * 2;            // 1.18 MB
  unsigned short* w2sw = (unsigned short*)(ws + o); o += (size_t)CDIM * 512 * 2;             // 0.59 MB
  unsigned short* asw  = (unsigned short*)(ws + o); o += (size_t)FEATURE_DIM * ASPECT_OUT * 2;
  float* adense        = (float*)(ws + o);          o += (size_t)FEATURE_DIM * ASPECT_OUT * 4;

  hipMemsetAsync(adense, 0, (size_t)FEATURE_DIM * ASPECT_OUT * sizeof(float), stream);
  aspect_scatter<<<12, 256, 0, stream>>>(w_off, w_def, w_err, w_inf, adense);

  // fragments = (K/32)*(N/16), threads = frags*32
  swizzle_kernel<<<(40 * 32 + 255) / 256, 256, 0, stream>>>(adense, asw, FEATURE_DIM, ASPECT_OUT);
  swizzle_kernel<<<(1152 * 32 + 255) / 256, 256, 0, stream>>>(W1, w1sw, CDIM, 1024);
  swizzle_kernel<<<(576 * 32 + 255) / 256, 256, 0, stream>>>(W2, w2sw, CDIM, 512);

  proj_kernel<<<(BATCH * SEQ) / 128, 256, 0, stream>>>(inputs, asw, b_off, b_def, b_err, b_inf, xbuf);
  gru_kernel<<<BATCH / BT, 256, 0, stream>>>(xbuf, w1sw, w2sw, b1, b2, (float*)d_out);
}
